// Mamba2Multitask_61211873902955
// MI455X (gfx1250) — matrix operand order repaired and checked by reading, not compiled, not run on hardware
//
#include <hip/hip_runtime.h>
#include <math.h>

// ---------------------------------------------------------------------------
// Model constants (from reference)
// ---------------------------------------------------------------------------
namespace {

constexpr int Bq   = 4;
constexpr int Lq   = 1024;
constexpr int INq  = 46;
constexpr int Dq   = 256;
constexpr int DSq  = 64;
constexpr int Kq   = 4;     // conv taps
constexpr int NLq  = 4;
constexpr int DIq  = 512;
constexpr int HDq  = 64;
constexpr int Hq   = 8;
constexpr int XBCq = 640;   // DI + 2*DS
constexpr int DIPq = 1160;  // 2*DI + 2*DS + H
constexpr int DFFq = 512;
constexpr int AHq  = 4;
constexpr int AHDq = 64;

typedef _Float16 h16;
typedef __attribute__((ext_vector_type(8)))  _Float16 v8h;
typedef __attribute__((ext_vector_type(16))) _Float16 v16h;
typedef __attribute__((ext_vector_type(8)))  float    v8f;
typedef __attribute__((ext_vector_type(4)))  unsigned u32x4;
typedef __attribute__((ext_vector_type(8)))  int      i32x8;
typedef __attribute__((ext_vector_type(4)))  int      i32x4;

__device__ inline v8f wmma_step(v16h a, v16h b, v8f c) {
  return __builtin_amdgcn_wmma_f32_16x16x32_f16(
      false, a, false, b, (short)0, c, false, false);
}

union FragU { v16h v; unsigned u[8]; };

// A fragment: LDS tile row-major [m][k].
__device__ inline v16h frag_a(const h16* lds, int row0, int kstep, int ldk) {
  const int lane = threadIdx.x & 31;
  const int hiq  = lane >> 4;
  const int m    = lane & 15;
  const unsigned* base = (const unsigned*)(lds + (size_t)(row0 + m) * ldk + kstep);
  FragU r;
#pragma unroll
  for (int v = 0; v < 8; ++v) {
    const int kd = v + 8 * hiq;
    r.u[v] = base[kd];
  }
  return r.v;
}

// B fragment: LDS tile transposed [n][k].
__device__ inline v16h frag_b(const h16* lds, int col0, int kstep, int ldk) {
  const int lane = threadIdx.x & 31;
  const int hiq  = lane >> 4;
  const int n    = lane & 15;
  const unsigned* base =
      (const unsigned*)(lds + (size_t)(col0 + n) * ldk + kstep + 16 * hiq);
  FragU r;
#pragma unroll
  for (int v = 0; v < 8; ++v) r.u[v] = base[v];
  return r.v;
}

// Async global -> LDS copy of 16 bytes (GLOBAL_LOAD_ASYNC_TO_LDS_B128).
__device__ inline void async_b128(const h16* gsrc, h16* ldst) {
  const unsigned loff = (unsigned)(uintptr_t)ldst;  // LDS aperture: addr[31:0]
  asm volatile("global_load_async_to_lds_b128 %0, %1, off"
               :: "v"(loff), "v"(gsrc) : "memory");
}
__device__ inline void async_wait0() {
  asm volatile("s_wait_asynccnt 0x0" ::: "memory");
}
__device__ inline void async_wait1() {  // allow 1 newest copy in flight
  asm volatile("s_wait_asynccnt 0x1" ::: "memory");
}

// Tensor Data Mover: one descriptor moves a whole 2-D f16 tile to LDS.
// D# per CDNA5 ISA (2-D: groups 2/3 zero).
__device__ inline void tdm_load_2d_f16(const h16* gsrc, h16* ldst,
                                       unsigned tile_w, unsigned tile_h,
                                       unsigned row_stride,
                                       unsigned tens_w, unsigned tens_h) {
  const unsigned lds_addr = (unsigned)(uintptr_t)ldst;
  const unsigned long long ga = (unsigned long long)(uintptr_t)gsrc;
  u32x4 g0;
  g0[0] = 1u;                                            // count=1 (valid user D#)
  g0[1] = lds_addr;                                      // lds_addr
  g0[2] = (unsigned)(ga & 0xffffffffu);                  // global_addr[31:0]
  g0[3] = (unsigned)((ga >> 32) & 0x01ffffffu)           // global_addr[56:32]
        | (2u << 30);                                    // type=2 ("image")
  i32x8 g1;
  g1[0] = (int)(1u << 16);                               // data_size=1 -> 2 bytes
  g1[1] = (int)((tens_w & 0xffffu) << 16);               // tensor_dim0[15:0]
  g1[2] = (int)((tens_w >> 16) | ((tens_h & 0xffffu) << 16));  // dim0 hi | dim1 lo
  g1[3] = (int)((tens_h >> 16) | (tile_w << 16));        // dim1 hi | tile_dim0
  g1[4] = (int)tile_h;                                   // tile_dim1 (tile_dim2=0)
  g1[5] = (int)row_stride;                               // tensor_dim0_stride[31:0]
  g1[6] = 0;                                             // stride hi / dim1_stride lo
  g1[7] = 0;
  const i32x4 gz = {0, 0, 0, 0};
#if __clang_major__ >= 23
  const i32x8 gz8 = {0, 0, 0, 0, 0, 0, 0, 0};
  __builtin_amdgcn_tensor_load_to_lds(g0, g1, gz, gz, gz8, 0);
#else
  __builtin_amdgcn_tensor_load_to_lds(g0, g1, gz, gz, 0);
#endif
}

__device__ inline float sigmoidf_(float x) { return 1.f / (1.f + __expf(-x)); }
__device__ inline float siluf_(float x)    { return x * sigmoidf_(x); }
__device__ inline float softplusf_(float x){ return x > 20.f ? x : log1pf(__expf(x)); }
__device__ inline float geluf_(float x)    { return 0.5f * x * (1.f + erff(x * 0.70710678118f)); }

#define V8F_ZERO {0.f,0.f,0.f,0.f,0.f,0.f,0.f,0.f}

// ---------------------------------------------------------------------------
// Weight convert + transpose: W[k][n] f32 -> WT[n][k] f16
// ---------------------------------------------------------------------------
__global__ __launch_bounds__(256) void k_wt(
    const float* __restrict__ W, h16* __restrict__ WT, int Kd, int N) {
  const int i = blockIdx.x * 256 + threadIdx.x;
  if (i >= Kd * N) return;
  const int k = i / N, n = i - k * N;
  WT[(size_t)n * Kd + k] = (h16)W[i];
}

// ---------------------------------------------------------------------------
// Embedding: h = x @ in_w + in_b + pos  (f32 residual stream)
// ---------------------------------------------------------------------------
__global__ __launch_bounds__(256) void k_embed(
    const float* __restrict__ x, const float* __restrict__ in_w,
    const float* __restrict__ in_b, const float* __restrict__ pos,
    float* __restrict__ hout) {
  __shared__ float xr[INq];
  const int row = blockIdx.x;
  const int l   = row & (Lq - 1);
  if (threadIdx.x < INq) xr[threadIdx.x] = x[(size_t)row * INq + threadIdx.x];
  __syncthreads();
  const int d = threadIdx.x;
  float s = in_b[d];
  for (int i = 0; i < INq; ++i) s += xr[i] * in_w[i * Dq + d];
  hout[(size_t)row * Dq + d] = s + pos[l * Dq + d];
}

// ---------------------------------------------------------------------------
// LayerNorm over D=256 -> f16 output (GEMM A operand)
// ---------------------------------------------------------------------------
__global__ __launch_bounds__(256) void k_ln(
    const float* __restrict__ in, const float* __restrict__ w,
    const float* __restrict__ b, h16* __restrict__ out) {
  __shared__ float red[256];
  const int row = blockIdx.x;
  const float v = in[(size_t)row * Dq + threadIdx.x];
  red[threadIdx.x] = v;
  __syncthreads();
  for (int s = 128; s > 0; s >>= 1) {
    if (threadIdx.x < s) red[threadIdx.x] += red[threadIdx.x + s];
    __syncthreads();
  }
  const float mean = red[0] * (1.f / Dq);
  __syncthreads();
  const float dv = v - mean;
  red[threadIdx.x] = dv * dv;
  __syncthreads();
  for (int s = 128; s > 0; s >>= 1) {
    if (threadIdx.x < s) red[threadIdx.x] += red[threadIdx.x + s];
    __syncthreads();
  }
  const float var = red[0] * (1.f / Dq);
  out[(size_t)row * Dq + threadIdx.x] =
      (h16)(dv * rsqrtf(var + 1e-5f) * w[threadIdx.x] + b[threadIdx.x]);
}

// ---------------------------------------------------------------------------
// WMMA GEMM, double-buffered. A tile via per-lane async b128 copies; W tile
// via one Tensor-Data-Mover descriptor per workgroup (wave 0 issues, TENSORcnt).
// ---------------------------------------------------------------------------
template <bool HB, bool HA, bool OF32, bool EDGE>
__global__ __launch_bounds__(256) void k_gemm(
    const h16* __restrict__ A, const h16* __restrict__ WT,
    const float* __restrict__ bias, const float* __restrict__ addsrc,
    float* __restrict__ Cf, h16* __restrict__ Ch, int M, int N, int Kd) {
  __shared__ __align__(16) h16 As[2][64 * 32];   // [m][k]
  __shared__ __align__(16) h16 Ws[2][64 * 32];   // [n][k]
  const int bm = blockIdx.y * 64, bn = blockIdx.x * 64;
  const int w = threadIdx.x >> 5, lane = threadIdx.x & 31;
  const int wr = (w >> 1) * 16, wc = (w & 1) * 32;
  // per-thread 16B copy slot for the A tile
  const int crow = threadIdx.x >> 2;
  const int cch  = (threadIdx.x & 3) * 8;
  const h16* gA = A  + (size_t)(bm + crow) * Kd + cch;
  const h16* gW = WT + (size_t)bn * Kd;             // tile-origin for TDM
  const unsigned trows = (unsigned)(N - bn);        // OOB rows zero-fill (EDGE)
  const int nsteps = Kd >> 5;
  // prologue: step-0 tiles
  async_b128(gA, &As[0][(crow << 5) + cch]);
  if (threadIdx.x < 32)
    tdm_load_2d_f16(gW, &Ws[0][0], 32, 64, (unsigned)Kd, (unsigned)Kd, trows);
  v8f acc0 = V8F_ZERO, acc1 = V8F_ZERO;
  for (int i = 0; i < nsteps; ++i) {
    const int cur = i & 1;
    if (i + 1 < nsteps) {
      // prefetch next K-tile into the other buffer, overlap with compute
      async_b128(gA + (i + 1) * 32, &As[1 - cur][(crow << 5) + cch]);
      if (threadIdx.x < 32) {
        tdm_load_2d_f16(gW + (i + 1) * 32, &Ws[1 - cur][0], 32, 64,
                        (unsigned)Kd, (unsigned)Kd, trows);
        __builtin_amdgcn_s_wait_tensorcnt(1);
      }
      async_wait1();
    } else {
      if (threadIdx.x < 32) __builtin_amdgcn_s_wait_tensorcnt(0);
      async_wait0();
    }
    __syncthreads();
    const v16h af = frag_a(&As[cur][0], wr, 0, 32);
    const v16h b0 = frag_b(&Ws[cur][0], wc, 0, 32);
    const v16h b1 = frag_b(&Ws[cur][0], wc + 16, 0, 32);
    acc0 = wmma_step(af, b0, acc0);
    acc1 = wmma_step(af, b1, acc1);
    __syncthreads();
  }
  const int hiq = lane >> 4, n = lane & 15;
#pragma unroll
  for (int r = 0; r < 8; ++r) {
    const int row = bm + wr + r + 8 * hiq;
    const int c0 = bn + wc + n, c1 = c0 + 16;
    float v0 = acc0[r], v1 = acc1[r];
    if (HB) { v0 += bias[c0]; v1 += bias[c1]; }
    if (HA) {
      v0 += addsrc[(size_t)row * N + c0];
      v1 += addsrc[(size_t)row * N + c1];
    }
    if (OF32) {
      if (!EDGE || c0 < N) Cf[(size_t)row * N + c0] = v0;
      if (!EDGE || c1 < N) Cf[(size_t)row * N + c1] = v1;
    } else {
      if (!EDGE || c0 < N) Ch[(size_t)row * N + c0] = (h16)v0;
      if (!EDGE || c1 < N) Ch[(size_t)row * N + c1] = (h16)v1;
    }
  }
}

// ---------------------------------------------------------------------------
// Depthwise causal conv(K=4) + SiLU -> xbch (f16); softplus(dt), a  (f32)
// ---------------------------------------------------------------------------
__global__ __launch_bounds__(256) void k_conv(
    const float* __restrict__ zx, const float* __restrict__ cw,
    const float* __restrict__ cb, const float* __restrict__ dt_bias,
    const float* __restrict__ A_log, h16* __restrict__ xbch,
    float* __restrict__ dtb, float* __restrict__ ab) {
  const int row = blockIdx.x;
  const int l   = row & (Lq - 1);
  const size_t base = (size_t)row * DIPq;
  for (int c = threadIdx.x; c < XBCq; c += 256) {
    float s = cb[c];
#pragma unroll
    for (int k = 0; k < Kq; ++k) {
      const int dl = l + k - (Kq - 1);
      if (dl >= 0)
        s += zx[(size_t)(row + k - (Kq - 1)) * DIPq + DIq + c] * cw[k * XBCq + c];
    }
    xbch[(size_t)row * XBCq + c] = (h16)siluf_(s);
  }
  if (threadIdx.x < Hq) {
    const float d0 = zx[base + DIq + XBCq + threadIdx.x] + dt_bias[threadIdx.x];
    const float sp = softplusf_(d0);
    dtb[(size_t)row * Hq + threadIdx.x] = sp;
    ab[(size_t)row * Hq + threadIdx.x]  = sp * -__expf(A_log[threadIdx.x]);
  }
}

__global__ void k_cumsum(const float* __restrict__ ab, float* __restrict__ cA) {
  const int t = threadIdx.x;
  if (t >= Bq * Hq) return;
  const int b = t / Hq, hh = t % Hq;
  float run = 0.f;
  for (int l = 0; l < Lq; ++l) {
    run += ab[((size_t)(b * Lq + l)) * Hq + hh];
    cA[((size_t)(b * Hq + hh)) * Lq + l] = run;
  }
}

// ---------------------------------------------------------------------------
// Flash-style SSM (causal, per (b,h,t-block64)), async-staged f16 tiles
// ---------------------------------------------------------------------------
__global__ __launch_bounds__(256) void k_ssm(
    const h16* __restrict__ xbch, const float* __restrict__ dtb,
    const float* __restrict__ cA, const float* __restrict__ Dh,
    float* __restrict__ yout) {
  __shared__ __align__(16) h16 Ct[64 * 64];   // [t][n]
  __shared__ __align__(16) h16 Bs[64 * 64];   // [s][n]
  __shared__ __align__(16) h16 Xt[64 * 64];   // [p][s]
  __shared__ __align__(16) h16 St[64 * 64];   // [t][s]
  __shared__ float cAt[64], cAs[64], dts[64];
  const int tb = blockIdx.x & 15;
  const int bh = blockIdx.x >> 4;
  const int b = bh >> 3, hh = bh & 7;
  const int w = threadIdx.x >> 5, lane = threadIdx.x & 31;
  const int tr = (w & 3) * 16, pc = (w >> 2) * 32;
  const int hiq = lane >> 4, nn = lane & 15;
  const float dval = Dh[hh];

  // stage C tile once via async copies
  for (int i = threadIdx.x; i < 512; i += 256) {
    const int t = i >> 3, c8 = (i & 7) * 8;
    async_b128(xbch + ((size_t)(b * Lq + tb * 64 + t)) * XBCq + DIq + DSq + c8,
               Ct + (t << 6) + c8);
  }
  if (threadIdx.x < 64)
    cAt[threadIdx.x] = cA[((size_t)(b * Hq + hh)) * Lq + tb * 64 + threadIdx.x];

  v8f y0 = V8F_ZERO, y1 = V8F_ZERO;
  for (int sb = 0; sb <= tb; ++sb) {
    __syncthreads();
    for (int i = threadIdx.x; i < 512; i += 256) {
      const int s = i >> 3, c8 = (i & 7) * 8;
      const size_t rs = (size_t)(b * Lq + sb * 64 + s);
      async_b128(xbch + rs * XBCq + DIq + c8, Bs + (s << 6) + c8);
      const v8h xv = *(const v8h*)(xbch + rs * XBCq + hh * HDq + c8);
#pragma unroll
      for (int j = 0; j < 8; ++j) Xt[((c8 + j) << 6) + s] = xv[j];
    }
    if (threadIdx.x < 64) {
      const int sg = sb * 64 + threadIdx.x;
      cAs[threadIdx.x] = cA[((size_t)(b * Hq + hh)) * Lq + sg];
      dts[threadIdx.x] = dtb[((size_t)(b * Lq + sg)) * Hq + hh];
    }
    async_wait0();
    __syncthreads();
    v8f cb0 = V8F_ZERO, cb1 = V8F_ZERO;
    const int sc = pc;
#pragma unroll
    for (int ks = 0; ks < 64; ks += 32) {
      const v16h a  = frag_a(Ct, tr, ks, 64);
      const v16h b0 = frag_b(Bs, sc, ks, 64);
      const v16h b1 = frag_b(Bs, sc + 16, ks, 64);
      cb0 = wmma_step(a, b0, cb0);
      cb1 = wmma_step(a, b1, cb1);
    }
#pragma unroll
    for (int r = 0; r < 8; ++r) {
      const int tl = tr + r + 8 * hiq;
      const int tg = tb * 64 + tl;
      const int sl0 = sc + nn, sl1 = sl0 + 16;
      const int sg0 = sb * 64 + sl0, sg1 = sb * 64 + sl1;
      const float v0 = (sg0 <= tg) ? cb0[r] * dts[sl0] * __expf(cAt[tl] - cAs[sl0]) : 0.f;
      const float v1 = (sg1 <= tg) ? cb1[r] * dts[sl1] * __expf(cAt[tl] - cAs[sl1]) : 0.f;
      St[(tl << 6) + sl0] = (h16)v0;
      St[(tl << 6) + sl1] = (h16)v1;
    }
    __syncthreads();
#pragma unroll
    for (int ks = 0; ks < 64; ks += 32) {
      const v16h a  = frag_a(St, tr, ks, 64);
      const v16h b0 = frag_b(Xt, pc, ks, 64);
      const v16h b1 = frag_b(Xt, pc + 16, ks, 64);
      y0 = wmma_step(a, b0, y0);
      y1 = wmma_step(a, b1, y1);
    }
  }
#pragma unroll
  for (int r = 0; r < 8; ++r) {
    const int tl = tr + r + 8 * hiq;
    const int tg = tb * 64 + tl;
    const size_t rb = (size_t)(b * Lq + tg);
    const int p0 = pc + nn, p1 = p0 + 16;
    const float x0 = (float)xbch[rb * XBCq + hh * HDq + p0];
    const float x1 = (float)xbch[rb * XBCq + hh * HDq + p1];
    yout[rb * DIq + hh * HDq + p0] = y0[r] + dval * x0;
    yout[rb * DIq + hh * HDq + p1] = y1[r] + dval * x1;
  }
}

// ---------------------------------------------------------------------------
// Flash attention (non-causal), async-staged f16 tiles, f16 o out
// ---------------------------------------------------------------------------
__global__ __launch_bounds__(256) void k_attn(
    const h16* __restrict__ qkv, h16* __restrict__ obuf) {
  __shared__ __align__(16) h16 Qt[64 * 64];
  __shared__ __align__(16) h16 Kt[64 * 64];
  __shared__ __align__(16) h16 Vt[64 * 64];   // [p][s]
  __shared__ __align__(16) h16 Pt[64 * 64];
  __shared__ float Sc[64][64];
  __shared__ float red[64][4];
  __shared__ float mrun[64], lrun[64], alpha[64];
  const int tb = blockIdx.x & 15;
  const int bh = blockIdx.x >> 4;
  const int b = bh >> 2, hh = bh & 3;
  const int w = threadIdx.x >> 5, lane = threadIdx.x & 31;
  const int tr = (w & 3) * 16, pc = (w >> 2) * 32;
  const int hiq = lane >> 4, nn = lane & 15;
  const int jrow = threadIdx.x >> 2, qq = threadIdx.x & 3;

  for (int i = threadIdx.x; i < 512; i += 256) {
    const int t = i >> 3, c8 = (i & 7) * 8;
    async_b128(qkv + ((size_t)(b * Lq + tb * 64 + t)) * (3 * Dq) + hh * AHDq + c8,
               Qt + (t << 6) + c8);
  }
  if (threadIdx.x < 64) { mrun[threadIdx.x] = -3.0e38f; lrun[threadIdx.x] = 0.f; }

  v8f y0 = V8F_ZERO, y1 = V8F_ZERO;
  for (int sb = 0; sb < Lq / 64; ++sb) {
    __syncthreads();
    for (int i = threadIdx.x; i < 512; i += 256) {
      const int s = i >> 3, c8 = (i & 7) * 8;
      const size_t rs = (size_t)(b * Lq + sb * 64 + s);
      async_b128(qkv + rs * (3 * Dq) + Dq + hh * AHDq + c8, Kt + (s << 6) + c8);
      const v8h vv = *(const v8h*)(qkv + rs * (3 * Dq) + 2 * Dq + hh * AHDq + c8);
#pragma unroll
      for (int j = 0; j < 8; ++j) Vt[((c8 + j) << 6) + s] = vv[j];
    }
    async_wait0();
    __syncthreads();
    v8f s0 = V8F_ZERO, s1 = V8F_ZERO;
    const int sc = pc;
#pragma unroll
    for (int ks = 0; ks < 64; ks += 32) {
      const v16h a  = frag_a(Qt, tr, ks, 64);
      const v16h b0 = frag_b(Kt, sc, ks, 64);
      const v16h b1 = frag_b(Kt, sc + 16, ks, 64);
      s0 = wmma_step(a, b0, s0);
      s1 = wmma_step(a, b1, s1);
    }
#pragma unroll
    for (int r = 0; r < 8; ++r) {
      const int tl = tr + r + 8 * hiq;
      Sc[tl][sc + nn]      = s0[r] * 0.125f;
      Sc[tl][sc + nn + 16] = s1[r] * 0.125f;
    }
    __syncthreads();
    float pm = -3.0e38f;
    for (int c = 0; c < 16; ++c) pm = fmaxf(pm, Sc[jrow][qq * 16 + c]);
    red[jrow][qq] = pm;
    __syncthreads();
    if (qq == 0) {
      float mnew = fmaxf(fmaxf(red[jrow][0], red[jrow][1]),
                         fmaxf(red[jrow][2], red[jrow][3]));
      mnew = fmaxf(mnew, mrun[jrow]);
      alpha[jrow] = __expf(mrun[jrow] - mnew);
      mrun[jrow]  = mnew;
    }
    __syncthreads();
    float ps = 0.f;
    for (int c = 0; c < 16; ++c) {
      const float e = __expf(Sc[jrow][qq * 16 + c] - mrun[jrow]);
      Pt[(jrow << 6) + qq * 16 + c] = (h16)e;
      ps += e;
    }
    red[jrow][qq] = ps;
    __syncthreads();
    if (qq == 0)
      lrun[jrow] = lrun[jrow] * alpha[jrow] +
                   red[jrow][0] + red[jrow][1] + red[jrow][2] + red[jrow][3];
#pragma unroll
    for (int r = 0; r < 8; ++r) {
      const int tl = tr + r + 8 * hiq;
      const float al = alpha[tl];
      y0[r] = y0[r] * al;
      y1[r] = y1[r] * al;
    }
#pragma unroll
    for (int ks = 0; ks < 64; ks += 32) {
      const v16h a  = frag_a(Pt, tr, ks, 64);
      const v16h b0 = frag_b(Vt, pc, ks, 64);
      const v16h b1 = frag_b(Vt, pc + 16, ks, 64);
      y0 = wmma_step(a, b0, y0);
      y1 = wmma_step(a, b1, y1);
    }
  }
  __syncthreads();
#pragma unroll
  for (int r = 0; r < 8; ++r) {
    const int tl = tr + r + 8 * hiq;
    const int tg = tb * 64 + tl;
    const float inv = 1.f / lrun[tl];
    const size_t rb = (size_t)(b * Lq + tg);
    obuf[rb * Dq + hh * AHDq + pc + nn]      = (h16)(y0[r] * inv);
    obuf[rb * Dq + hh * AHDq + pc + nn + 16] = (h16)(y1[r] * inv);
  }
}

// y = y * silu(z); RMS-normalize over DI; * norm_w -> f16
__global__ __launch_bounds__(256) void k_gate_rms(
    const float* __restrict__ y, const float* __restrict__ zx,
    const float* __restrict__ nw, h16* __restrict__ out) {
  __shared__ float buf[DIq];
  __shared__ float red[256];
  const int row = blockIdx.x;
  float ss = 0.f;
  for (int c = threadIdx.x; c < DIq; c += 256) {
    const float z = zx[(size_t)row * DIPq + c];
    const float v = y[(size_t)row * DIq + c] * siluf_(z);
    buf[c] = v;
    ss += v * v;
  }
  red[threadIdx.x] = ss;
  __syncthreads();
  for (int s = 128; s > 0; s >>= 1) {
    if (threadIdx.x < s) red[threadIdx.x] += red[threadIdx.x + s];
    __syncthreads();
  }
  const float scale = rsqrtf(red[0] * (1.f / DIq) + 1e-5f);
  for (int c = threadIdx.x; c < DIq; c += 256)
    out[(size_t)row * DIq + c] = (h16)(buf[c] * scale * nw[c]);
}

// GEGLU: gelu(x1) * x2   (f16 in/out)
__global__ __launch_bounds__(256) void k_geglu(
    const h16* __restrict__ in, h16* __restrict__ out) {
  const int row = blockIdx.x;
  for (int c = threadIdx.x; c < DFFq; c += 256) {
    const float x1 = (float)in[(size_t)row * (2 * DFFq) + c];
    const float x2 = (float)in[(size_t)row * (2 * DFFq) + DFFq + c];
    out[(size_t)row * DFFq + c] = (h16)(geluf_(x1) * x2);
  }
}

// mean over L (reads final LN in f16)
__global__ __launch_bounds__(256) void k_mean(
    const h16* __restrict__ in, float* __restrict__ hm) {
  const int b = blockIdx.x, d = threadIdx.x;
  float s = 0.f;
  for (int l = 0; l < Lq; ++l) s += (float)in[((size_t)(b * Lq + l)) * Dq + d];
  hm[b * Dq + d] = s * (1.f / Lq);
}

__global__ void k_heads(
    const float* __restrict__ hm, const float* __restrict__ dw,
    const float* __restrict__ db, const float* __restrict__ rw,
    const float* __restrict__ rb, float* __restrict__ out) {
  const int t = threadIdx.x;
  if (t >= 12) return;
  const int b = t / 3, j = t % 3;
  float sd = db[j], sr = rb[j];
  for (int d = 0; d < Dq; ++d) {
    const float h = hm[b * Dq + d];
    sd += h * dw[d * 3 + j];
    sr += h * rw[d * 3 + j];
  }
  out[t]      = sd;
  out[12 + t] = sr;
}

}  // anonymous namespace

// ---------------------------------------------------------------------------
// Orchestration
// ---------------------------------------------------------------------------
extern "C" void kernel_launch(void* const* d_in, const int* in_sizes, int n_in,
                              void* d_out, int out_size, void* d_ws, size_t ws_size,
                              hipStream_t stream) {
  (void)in_sizes; (void)n_in; (void)out_size; (void)ws_size;
  const float* x        = (const float*)d_in[0];
  const float* pos      = (const float*)d_in[1];
  const float* in_w     = (const float*)d_in[2];
  const float* in_b     = (const float*)d_in[3];
  const float* ln1_w    = (const float*)d_in[4];
  const float* ln1_b    = (const float*)d_in[5];
  const float* ssm_in_w = (const float*)d_in[6];
  const float* conv_w   = (const float*)d_in[7];
  const float* conv_b   = (const float*)d_in[8];
  const float* dt_bias  = (const float*)d_in[9];
  const float* A_log    = (const float*)d_in[10];
  const float* ssm_D    = (const float*)d_in[11];
  const float* norm_wi  = (const float*)d_in[12];
  const float* out_wi   = (const float*)d_in[13];
  const float* ln2_w    = (const float*)d_in[14];
  const float* ln2_b    = (const float*)d_in[15];
  const float* fc1_w    = (const float*)d_in[16];
  const float* fc1_b    = (const float*)d_in[17];
  const float* fc2_w    = (const float*)d_in[18];
  const float* fc2_b    = (const float*)d_in[19];
  const float* aln_w    = (const float*)d_in[20];
  const float* aln_b    = (const float*)d_in[21];
  const float* qkv_w    = (const float*)d_in[22];
  const float* qkv_b    = (const float*)d_in[23];
  const float* aout_w   = (const float*)d_in[24];
  const float* aout_b   = (const float*)d_in[25];
  const float* norm_w   = (const float*)d_in[26];
  const float* norm_b   = (const float*)d_in[27];
  const float* dir_w    = (const float*)d_in[28];
  const float* dir_b    = (const float*)d_in[29];
  const float* reg_w    = (const float*)d_in[30];
  const float* reg_b    = (const float*)d_in[31];

  const size_t ROWS = (size_t)Bq * Lq;  // 4096
  // ---- f32 region ----
  float* wsf  = (float*)d_ws;
  float* h    = wsf; wsf += ROWS * Dq;
  float* big  = wsf; wsf += ROWS * DIPq;
  float* ybuf = wsf; wsf += ROWS * DIq;
  float* dtb  = wsf; wsf += ROWS * Hq;
  float* ab   = wsf; wsf += ROWS * Hq;
  float* cAb  = wsf; wsf += ROWS * Hq;
  float* hm   = wsf; wsf += (size_t)Bq * Dq;
  // ---- f16 region ----
  h16* ws16   = (h16*)wsf;
  h16* lnh    = ws16; ws16 += ROWS * Dq;
  h16* yb2h   = ws16; ws16 += ROWS * DIq;
  h16* obufh  = ws16; ws16 += ROWS * Dq;
  h16* qkvh   = ws16; ws16 += ROWS * 3 * Dq;
  h16* fc1h   = ws16; ws16 += ROWS * 2 * DFFq;
  h16* xbch   = ws16; ws16 += ROWS * XBCq;
  h16* wt_in  = ws16; ws16 += (size_t)NLq * DIPq * Dq;
  h16* wt_out = ws16; ws16 += (size_t)NLq * Dq * DIq;
  h16* wt_qkv = ws16; ws16 += (size_t)2 * 3 * Dq * Dq;
  h16* wt_ao  = ws16; ws16 += (size_t)2 * Dq * Dq;
  h16* wt_f1  = ws16; ws16 += (size_t)NLq * 2 * DFFq * Dq;
  h16* wt_f2  = ws16; ws16 += (size_t)NLq * Dq * DFFq;

  const dim3 blk(256);
  auto cvtw = [&](const float* W, h16* WT, int Kd, int N) {
    k_wt<<<dim3((unsigned)((Kd * N + 255) / 256)), blk, 0, stream>>>(W, WT, Kd, N);
  };
  for (int i = 0; i < NLq; ++i) {
    cvtw(ssm_in_w + (size_t)i * Dq * DIPq, wt_in  + (size_t)i * DIPq * Dq, Dq,  DIPq);
    cvtw(out_wi   + (size_t)i * DIq * Dq,  wt_out + (size_t)i * Dq * DIq,  DIq, Dq);
    cvtw(fc1_w    + (size_t)i * Dq * 2 * DFFq, wt_f1 + (size_t)i * 2 * DFFq * Dq, Dq, 2 * DFFq);
    cvtw(fc2_w    + (size_t)i * DFFq * Dq, wt_f2  + (size_t)i * Dq * DFFq, DFFq, Dq);
  }
  for (int j = 0; j < 2; ++j) {
    cvtw(qkv_w  + (size_t)j * Dq * 3 * Dq, wt_qkv + (size_t)j * 3 * Dq * Dq, Dq, 3 * Dq);
    cvtw(aout_w + (size_t)j * Dq * Dq,     wt_ao  + (size_t)j * Dq * Dq,     Dq, Dq);
  }

  k_embed<<<dim3((unsigned)ROWS), blk, 0, stream>>>(x, in_w, in_b, pos, h);

  for (int i = 0; i < NLq; ++i) {
    // ---- Mamba2 block ----
    k_ln<<<dim3((unsigned)ROWS), blk, 0, stream>>>(h, ln1_w + i * Dq, ln1_b + i * Dq, lnh);
    k_gemm<false, false, true, true>
        <<<dim3((DIPq + 63) / 64, (unsigned)(ROWS / 64)), blk, 0, stream>>>(
        lnh, wt_in + (size_t)i * DIPq * Dq, nullptr, nullptr, big, nullptr,
        (int)ROWS, DIPq, Dq);
    k_conv<<<dim3((unsigned)ROWS), blk, 0, stream>>>(
        big, conv_w + (size_t)i * Kq * XBCq, conv_b + i * XBCq,
        dt_bias + i * Hq, A_log + i * Hq, xbch, dtb, ab);
    k_cumsum<<<dim3(1), dim3(32), 0, stream>>>(ab, cAb);
    k_ssm<<<dim3(Bq * Hq * (Lq / 64)), blk, 0, stream>>>(xbch, dtb, cAb, ssm_D + i * Hq, ybuf);
    k_gate_rms<<<dim3((unsigned)ROWS), blk, 0, stream>>>(ybuf, big, norm_wi + i * DIq, yb2h);
    k_gemm<false, true, true, false>
        <<<dim3(Dq / 64, (unsigned)(ROWS / 64)), blk, 0, stream>>>(
        yb2h, wt_out + (size_t)i * Dq * DIq, nullptr, h, h, nullptr,
        (int)ROWS, Dq, DIq);

    // ---- Attention (odd layers) ----
    if (i % 2 == 1) {
      const int j = i / 2;
      k_ln<<<dim3((unsigned)ROWS), blk, 0, stream>>>(h, aln_w + j * Dq, aln_b + j * Dq, lnh);
      k_gemm<true, false, false, false>
          <<<dim3((3 * Dq) / 64, (unsigned)(ROWS / 64)), blk, 0, stream>>>(
          lnh, wt_qkv + (size_t)j * 3 * Dq * Dq, qkv_b + j * 3 * Dq, nullptr,
          nullptr, qkvh, (int)ROWS, 3 * Dq, Dq);
      k_attn<<<dim3(Bq * AHq * (Lq / 64)), blk, 0, stream>>>(qkvh, obufh);
      k_gemm<true, true, true, false>
          <<<dim3(Dq / 64, (unsigned)(ROWS / 64)), blk, 0, stream>>>(
          obufh, wt_ao + (size_t)j * Dq * Dq, aout_b + j * Dq, h, h, nullptr,
          (int)ROWS, Dq, Dq);
    }

    // ---- gated MLP ----
    k_ln<<<dim3((unsigned)ROWS), blk, 0, stream>>>(h, ln2_w + i * Dq, ln2_b + i * Dq, lnh);
    k_gemm<true, false, false, false>
        <<<dim3((2 * DFFq) / 64, (unsigned)(ROWS / 64)), blk, 0, stream>>>(
        lnh, wt_f1 + (size_t)i * 2 * DFFq * Dq, fc1_b + i * 2 * DFFq, nullptr,
        nullptr, fc1h, (int)ROWS, 2 * DFFq, Dq);
    k_geglu<<<dim3((unsigned)ROWS), blk, 0, stream>>>(fc1h, yb2h);
    k_gemm<true, true, true, false>
        <<<dim3(Dq / 64, (unsigned)(ROWS / 64)), blk, 0, stream>>>(
        yb2h, wt_f2 + (size_t)i * Dq * DFFq, fc2_b + i * Dq, h, h, nullptr,
        (int)ROWS, Dq, DFFq);
  }

  k_ln<<<dim3((unsigned)ROWS), blk, 0, stream>>>(h, norm_w, norm_b, lnh);
  k_mean<<<dim3(Bq), blk, 0, stream>>>(lnh, hm);
  k_heads<<<dim3(1), dim3(32), 0, stream>>>(hm, dir_w, dir_b, reg_w, reg_b, (float*)d_out);
}
